// MotionCompleteModel_62792421867850
// MI455X (gfx1250) — matrix operand order repaired and checked by reading, not compiled, not run on hardware
//
#include <hip/hip_runtime.h>

typedef __attribute__((ext_vector_type(16))) __bf16 v16bf;
typedef __attribute__((ext_vector_type(8)))  float  v8f;
typedef int v4i __attribute__((__vector_size__(16)));

#define BN_EPS 1e-4f

#if defined(__HIP_DEVICE_COMPILE__) && \
    __has_builtin(__builtin_amdgcn_global_load_async_to_lds_b128) && \
    __has_builtin(__builtin_amdgcn_s_wait_asynccnt)
#define HAVE_ASYNC_LDS 1
#else
#define HAVE_ASYNC_LDS 0
#endif

#if defined(__HIP_DEVICE_COMPILE__) && \
    __has_builtin(__builtin_amdgcn_tensor_load_to_lds) && \
    __has_builtin(__builtin_amdgcn_s_wait_tensorcnt)
#define HAVE_TDM 1
#else
#define HAVE_TDM 0
#endif

#if HAVE_TDM
typedef unsigned tdu4 __attribute__((ext_vector_type(4)));
typedef int      tdi8 __attribute__((ext_vector_type(8)));
typedef int      tdi4 __attribute__((ext_vector_type(4)));
#endif

static inline int nblk(long n) { return (int)((n + 255) / 256); }

// ------------------------------------------------------------------
// Elementwise: fused BN + ReLU + mask, f32 -> bf16 (with channel pad)
// ------------------------------------------------------------------
__global__ void k_bnrelu_bf16(const float* __restrict__ x, int xs, int xo,
                              const float* __restrict__ g, const float* __restrict__ bb,
                              const float* __restrict__ mu, const float* __restrict__ vv,
                              const int* __restrict__ mask, __bf16* __restrict__ out,
                              int C, int CPAD, long nvox, int relu)
{
  long i = (long)blockIdx.x * blockDim.x + threadIdx.x;
  long total = nvox * CPAD;
  if (i >= total) return;
  int c = (int)(i % CPAD);
  long vx = i / CPAD;
  float val = 0.f;
  if (c < C) {
    val = x[vx * xs + xo + c];
    if (g) {
      float s = g[c] * rsqrtf(vv[c] + BN_EPS);
      val = val * s + (bb[c] - mu[c] * s);
    }
    if (relu) val = fmaxf(val, 0.f);
    val *= (float)mask[vx];
  }
  out[vx * CPAD + c] = (__bf16)val;
}

// ------------------------------------------------------------------
// 1x1 NetworkInNetwork on identity branch (f32, small FLOPs)
// ------------------------------------------------------------------
__global__ void k_nin(const float* __restrict__ x, int xs, int xo,
                      const float* __restrict__ w, float* __restrict__ out,
                      int os, int oo, int CI, int CO, long nvox)
{
  long i = (long)blockIdx.x * blockDim.x + threadIdx.x;
  if (i >= nvox * CO) return;
  int co = (int)(i % CO);
  long vx = i / CO;
  const float* xp = x + vx * xs + xo;
  float s = 0.f;
  for (int ci = 0; ci < CI; ++ci) s += xp[ci] * w[(long)ci * CO + co];
  out[vx * os + oo + co] = s;
}

// ------------------------------------------------------------------
// 2x2x2 mask max-pool (downsample active-site mask)
// ------------------------------------------------------------------
__global__ void k_downmask(const int* __restrict__ m, int* __restrict__ o,
                           int CX, int CY, int CZ)
{
  long i = (long)blockIdx.x * blockDim.x + threadIdx.x;
  long tot = 2L * CX * CY * CZ;
  if (i >= tot) return;
  int cz = (int)(i % CZ); long r = i / CZ;
  int cy = (int)(r % CY); r /= CY;
  int cx = (int)(r % CX); int b = (int)(r / CX);
  int X = 2 * CX, Y = 2 * CY, Z = 2 * CZ;
  int mv = 0;
  for (int a = 0; a < 2; a++)
    for (int c = 0; c < 2; c++)
      for (int d = 0; d < 2; d++) {
        long fi = ((((long)b * X + 2 * cx + a) * Y + 2 * cy + c) * Z + 2 * cz + d);
        mv |= (m[fi] != 0);
      }
  o[i] = mv;
}

// ------------------------------------------------------------------
// Head: BN+ReLU+mask then 16x3 linear + bias, *mask
// ------------------------------------------------------------------
__global__ void k_head(const float* __restrict__ x,
                       const float* __restrict__ g, const float* __restrict__ bb,
                       const float* __restrict__ mu, const float* __restrict__ vv,
                       const float* __restrict__ lw, const float* __restrict__ lb,
                       const int* __restrict__ mask, float* __restrict__ out, long nvox)
{
  long vx = (long)blockIdx.x * blockDim.x + threadIdx.x;
  if (vx >= nvox) return;
  float mv = (float)mask[vx];
  float a0 = lb[0], a1 = lb[1], a2 = lb[2];
  for (int c = 0; c < 16; c++) {
    float s = g[c] * rsqrtf(vv[c] + BN_EPS);
    float h = fmaxf(x[vx * 16 + c] * s + (bb[c] - mu[c] * s), 0.f) * mv;
    a0 += h * lw[c * 3 + 0];
    a1 += h * lw[c * 3 + 1];
    a2 += h * lw[c * 3 + 2];
  }
  out[vx * 3 + 0] = a0 * mv;
  out[vx * 3 + 1] = a1 * mv;
  out[vx * 3 + 2] = a2 * mv;
}

// ------------------------------------------------------------------
// Weight repack: f32 [tap][ci][co] -> bf16 [tap][ci/2][co][2]
// ------------------------------------------------------------------
__global__ void k_repack(const float* __restrict__ w, __bf16* __restrict__ o,
                         int ntin, int ntout, int ci, int cipad, int co)
{
  long i = (long)blockIdx.x * blockDim.x + threadIdx.x;
  long tot = (long)ntout * (cipad / 2) * co * 2;
  if (i >= tot) return;
  int r = (int)(i & 1);
  long q = i >> 1;
  int n = (int)(q % co); q /= co;
  int p = (int)(q % (cipad / 2));
  int t = (int)(q / (cipad / 2));
  int c = 2 * p + r;
  float val = (t < ntin && c < ci) ? w[((long)t * ci + c) * co + n] : 0.f;
  o[i] = (__bf16)val;
}

// ------------------------------------------------------------------
// Tap -> spatial offset decode
// ------------------------------------------------------------------
template<int KS, bool DEC>
__device__ __forceinline__ void dec_off(int t, int& dx, int& dy, int& dz)
{
  if (DEC || KS == 1) { dx = 0; dy = 0; dz = 0; }
  else if (KS == 3) {
    if (t >= 27) { dx = 1; dy = 1; dz = 1; }         // zero-weight pad tap
    else { dx = t / 9; dy = (t / 3) % 3; dz = t % 3; }
  } else { dx = (t >> 2) & 1; dy = (t >> 1) & 1; dz = t & 1; }
}

// ------------------------------------------------------------------
// Implicit-GEMM conv via V_WMMA_F32_16X16X32_BF16.
//   tile = 4x2x16 output voxels, 8 waves; M=16 voxels along z per wave,
//   N-loop over COUT/16 tiles, f32 accumulators. Interior tiles stage
//   the halo slab with one TENSOR_LOAD_TO_LDS (TDM 4D tile descriptor,
//   dim0-fastest == our ci-fastest LDS layout); border tiles use
//   GLOBAL_LOAD_ASYNC_TO_LDS_B128 with zero-filled halo. A-fragments
//   are ds_load_b128 pairs per the 16-bit A 16x32 striping. Epilogue
//   fuses the submanifold mask multiply (+ optional identity add).
//   DEC=true: stride-2 deconv; grid.z carries the 8 output parities.
// ------------------------------------------------------------------
template<int CIN, int COUT, int KS, int PAD, int STR, bool DEC>
__global__ __launch_bounds__(256)
void k_conv_wmma(const __bf16* __restrict__ in, const unsigned* __restrict__ wts,
                 const float* __restrict__ add, int as_, int ao,
                 const int* __restrict__ mask,
                 float* __restrict__ out, int os_, int oo,
                 int X, int Y, int Z, int OX, int OY, int OZ, int ZT)
{
  constexpr int TX = 4, TY = 2, TZ = 16;
  constexpr int IX = DEC ? TX : TX * STR + KS - STR;
  constexpr int IY = DEC ? TY : TY * STR + KS - STR;
  constexpr int IZ = DEC ? TZ : TZ * STR + KS - STR;
  constexpr int CK = (CIN < 32) ? CIN : 32;      // staged channel chunk
  constexpr int NCH = CIN / CK;
  constexpr int NT = DEC ? 1 : KS * KS * KS;
  constexpr int UNITS = NT * (CK / 16);          // K16-units per chunk
  constexpr int UP = (UNITS + 1) & ~1;           // padded to WMMA K=32 pairs
  constexpr int NTN = COUT / 16;
  constexpr int SITES = IX * IY * IZ;
  constexpr int SMW = SITES * (CK / 2);          // dwords (2 bf16 each)
  __shared__ __align__(16) unsigned sm[SMW];

  int zb = blockIdx.z;
  int b, zt, ptap = 0, px = 0, py = 0, pz = 0;
  if (DEC) {
    int r = zb;
    b = r / (ZT * 8); r %= (ZT * 8);
    ptap = r / ZT; zt = r % ZT;
    px = (ptap >> 2) & 1; py = (ptap >> 1) & 1; pz = ptap & 1;
  } else { b = zb / ZT; zt = zb % ZT; }

  int x0 = blockIdx.x * TX, y0 = blockIdx.y * TY, z0 = zt * TZ;
  int tid = threadIdx.x;
  int lane = tid & 31, wv = tid >> 5;
  int tx = wv >> 1, ty = wv & 1;
  int gl = lane >> 4, n16 = lane & 15;   // A: m=n16 ; B/C/D: n=n16

  if (tid == 0) __builtin_prefetch(wts, 0, 1);   // global_prefetch_b8

  v8f acc[NTN];
  #pragma unroll
  for (int j = 0; j < NTN; j++)
    #pragma unroll
    for (int e = 0; e < 8; e++) acc[j][e] = 0.f;

  int gx0 = x0 * STR - PAD, gy0 = y0 * STR - PAD, gz0 = z0 * STR - PAD;

  for (int chnk = 0; chnk < NCH; ++chnk) {
    __syncthreads();
#if HAVE_TDM
    bool interior = (gx0 >= 0) && (gx0 + IX <= X) &&
                    (gy0 >= 0) && (gy0 + IY <= Y) &&
                    (gz0 >= 0) && (gz0 + IZ <= Z);
    if (interior) {
      // ---- one TDM descriptor: 4D tile (CK x IZ x IY x IX), 2B elems ----
      if (wv == 0) {
        unsigned long long gaddr = (unsigned long long)in +
            ((((unsigned long long)(b * X + gx0) * Y + gy0) * Z + gz0) * CIN
             + (unsigned long long)chnk * CK) * 2ull;
        unsigned ldsa = (unsigned)(unsigned long long)(void*)&sm[0];
        unsigned long long s0 = (unsigned long long)CIN;          // dim0 stride
        unsigned long long s1 = (unsigned long long)Z * CIN;      // dim1 stride
        unsigned long long s2 = (unsigned long long)Y * Z * CIN;  // dim2 stride
        const unsigned BIGD = 1u << 30;                           // no OOB clip
        tdu4 g0; tdi8 g1; tdi4 g2, g3;
        g0[0] = 1u;                                   // count=1 (valid)
        g0[1] = ldsa;                                 // lds_addr
        g0[2] = (unsigned)(gaddr & 0xffffffffu);      // global_addr[31:0]
        g0[3] = (unsigned)((gaddr >> 32) & 0x1ffffffu) | (2u << 30); // type=2
        g1[0] = (int)(1u << 16);                      // data_size=2B
        g1[1] = (int)((BIGD & 0xffffu) << 16);        // tensor_dim0[15:0]
        g1[2] = (int)(((BIGD >> 16) & 0xffffu) | ((BIGD & 0xffffu) << 16));
        g1[3] = (int)(((BIGD >> 16) & 0xffffu) | ((unsigned)CK << 16)); // tile_dim0
        g1[4] = (int)((unsigned)IZ | ((unsigned)IY << 16)); // tile_dim1,2
        g1[5] = (int)(unsigned)(s0 & 0xffffffffu);          // stride0[31:0]
        g1[6] = (int)(((s0 >> 32) & 0xffffu) | ((s1 & 0xffffu) << 16));
        g1[7] = (int)(unsigned)((s1 >> 16) & 0xffffffffu);  // stride1[47:16]
        g2[0] = (int)BIGD;                                  // tensor_dim2
        g2[1] = (int)BIGD;                                  // tensor_dim3
        g2[2] = (int)(unsigned)(s2 & 0xffffffffu);          // stride2[31:0]
        g2[3] = (int)(((s2 >> 32) & 0xffffu) | ((unsigned)IX << 16)); // tile_dim3
        g3[0] = 0; g3[1] = 0; g3[2] = 0; g3[3] = 0;
#if defined(__clang_major__) && __clang_major__ >= 23
        tdi8 g4; for (int e = 0; e < 8; e++) g4[e] = 0;
        __builtin_amdgcn_tensor_load_to_lds(g0, g1, g2, g3, g4, 0);
#else
        __builtin_amdgcn_tensor_load_to_lds(g0, g1, g2, g3, 0);
#endif
        __builtin_amdgcn_s_wait_tensorcnt(0);
      }
    } else
#endif
#if HAVE_ASYNC_LDS
    {
      // ---- border: per-lane memory->LDS async b128 copies + zero fill ----
      constexpr int SM16 = SMW / 4;               // 16-byte units
      constexpr int CQ = CK / 8;                  // 16B units per site
      for (int d4 = tid; d4 < SM16; d4 += 256) {
        int cq = d4 % CQ;
        int site = d4 / CQ;
        int iz = site % IZ; int r2 = site / IZ;
        int iy = r2 % IY;   int ix = r2 / IY;
        int gx = gx0 + ix, gy = gy0 + iy, gz = gz0 + iz;
        if (gx >= 0 && gx < X && gy >= 0 && gy < Y && gz >= 0 && gz < Z) {
          long vix = ((long)(b * X + gx) * Y + gy) * Z + gz;
          const unsigned* gp = (const unsigned*)in + vix * (CIN / 2)
                               + (chnk * CK) / 2 + cq * 4;
          __builtin_amdgcn_global_load_async_to_lds_b128(
              (__attribute__((address_space(1))) v4i*)(void*)gp,
              (__attribute__((address_space(3))) v4i*)(void*)&sm[d4 * 4], 0, 0);
        } else {
          uint4 z4; z4.x = 0; z4.y = 0; z4.z = 0; z4.w = 0;
          *(uint4*)&sm[d4 * 4] = z4;              // ds_store_b128 zero fill
        }
      }
      __builtin_amdgcn_s_wait_asynccnt(0);
    }
#else
    {
      // ---- fallback staging: bf16 dwords via VGPRs ----
      for (int d = tid; d < SMW; d += 256) {
        int cpair = d % (CK / 2);
        int site  = d / (CK / 2);
        int iz = site % IZ; int r2 = site / IZ;
        int iy = r2 % IY;   int ix = r2 / IY;
        int gx = gx0 + ix, gy = gy0 + iy, gz = gz0 + iz;
        unsigned val = 0;
        if (gx >= 0 && gx < X && gy >= 0 && gy < Y && gz >= 0 && gz < Z) {
          long vix = ((long)(b * X + gx) * Y + gy) * Z + gz;
          val = ((const unsigned*)in)[vix * (CIN / 2) + (chnk * CK) / 2 + cpair];
        }
        sm[d] = val;
      }
    }
#endif
    __syncthreads();

    // ---- K loop: pairs of K16-units -> one K=32 WMMA per N-tile ----
    for (int u = 0; u < UP; u += 2) {
      int t0, c00, t1, c01;
      if (CK == 16) { t0 = u; c00 = 0; t1 = u + 1; c01 = 0; }
      else          { t0 = u >> 1; c00 = 0; t1 = u >> 1; c01 = 16; }

      int dx0, dy0, dz0, dx1, dy1, dz1;
      dec_off<KS, DEC>(t0, dx0, dy0, dz0);
      dec_off<KS, DEC>(t1, dx1, dy1, dz1);

      union { v16bf v; unsigned u32[8]; uint4 q[2]; } A, B;

      // A fragment: 16-bit A 16x32 striping -> two ds_load_b128
      {
        int m = n16;
        int s0 = ((tx * STR + dx0) * IY + (ty * STR + dy0)) * IZ + (m * STR + dz0);
        int s1 = ((tx * STR + dx1) * IY + (ty * STR + dy1)) * IZ + (m * STR + dz1);
        int cb0 = (c00 + 8 * gl) >> 1;   // multiple of 4
        int cb1 = (c01 + 8 * gl) >> 1;
        A.q[0] = *(const uint4*)&sm[s0 * (CK / 2) + cb0];
        A.q[1] = *(const uint4*)&sm[s1 * (CK / 2) + cb1];
      }

      // B fragment: elems 0-7 unit u ch 8*gl..+7, elems 8-15 unit u+1 ch 8*gl..+7
      int tb0 = DEC ? ptap : t0;
      int tb1 = DEC ? ptap : t1;
      int cbase0 = (chnk * CK + c00 + 8 * gl) >> 1;
      int cbase1 = (chnk * CK + c01 + 8 * gl) >> 1;
      #pragma unroll
      for (int nt = 0; nt < NTN; ++nt) {
        int co = nt * 16 + n16;
        #pragma unroll
        for (int i2 = 0; i2 < 8; i2++)
          B.u32[i2] = wts[((long)(i2 < 4 ? tb0 : tb1) * (CIN / 2) + (i2 < 4 ? cbase0 + i2 : cbase1 + i2 - 4)) * COUT + co];
        acc[nt] = __builtin_amdgcn_wmma_f32_16x16x32_bf16(
            false, A.v, false, B.v, (short)0, acc[nt], false, false);
      }
    }
  }

  // ---- epilogue: mask (+identity add), f32 store ----
  #pragma unroll
  for (int nt = 0; nt < NTN; ++nt) {
    #pragma unroll
    for (int r = 0; r < 8; r++) {
      int m = r + 8 * gl;             // C/D layout: VGPR r -> M=r / M=r+8
      int ox_, oy_, oz_;
      if (DEC) {
        if (z0 + m >= Z) continue;
        ox_ = 2 * (x0 + tx) + px; oy_ = 2 * (y0 + ty) + py; oz_ = 2 * (z0 + m) + pz;
      } else {
        oz_ = z0 + m;
        if (oz_ >= OZ) continue;
        ox_ = x0 + tx; oy_ = y0 + ty;
      }
      long vo = (((long)b * OX + ox_) * OY + oy_) * OZ + oz_;
      float mv = (float)mask[vo];
      float val = acc[nt][r] * mv;
      if (add) val += add[vo * as_ + ao + nt * 16 + n16];
      out[vo * os_ + oo + nt * 16 + n16] = val;
    }
  }
}

// ==================================================================
// Host side: param-tree parsing, weight repack, UNet orchestration
// ==================================================================
struct BNp { const float *g, *b, *mu, *v; };
struct BlkP { BNp bn1, bn2; const float *w1, *w2, *nin; };
struct LvlP { BlkP blk; BNp bnd, bnu; const float *wd, *wu; BlkP post; };

static void parse_level(int l, int& cur, void* const* din, LvlP* lv)
{
  auto nx = [&]() { return (const float*)din[cur++]; };
  auto bn = [&](BNp& o) { o.g = nx(); o.b = nx(); o.mu = nx(); o.v = nx(); };
  bn(lv[l].blk.bn1); bn(lv[l].blk.bn2);
  lv[l].blk.w1 = nx(); lv[l].blk.w2 = nx(); lv[l].blk.nin = nullptr;
  if (l < 3) {
    bn(lv[l].bnd); bn(lv[l].bnu);
    lv[l].wd = nx(); lv[l].wu = nx();
    parse_level(l + 1, cur, din, lv);
    bn(lv[l].post.bn1); bn(lv[l].post.bn2);
    lv[l].post.w1 = nx(); lv[l].post.w2 = nx(); lv[l].post.nin = nx();
  }
}

static void conv3_disp(int CI, int CO, const __bf16* in, const unsigned* w,
                       const float* add, int as, int ao, const int* mk,
                       float* out, int os, int oo, int S, hipStream_t st)
{
  int ZT = (S + 15) / 16;
  dim3 g(S / 4, S / 2, ZT * 2), blk(256, 1, 1);
#define LC3(ci, co) if (CI == ci && CO == co) { k_conv_wmma<ci, co, 3, 1, 1, false><<<g, blk, 0, st>>>(in, w, add, as, ao, mk, out, os, oo, S, S, S, S, S, S, ZT); return; }
  LC3(16, 16) LC3(32, 32) LC3(64, 64) LC3(128, 128)
  LC3(32, 16) LC3(64, 32) LC3(128, 64)
#undef LC3
}

static void down_disp(int CI, int CO, const __bf16* in, const unsigned* w,
                      const int* mk2, float* out, int os, int oo, int Sin, hipStream_t st)
{
  int So = Sin / 2;
  int ZT = (So + 15) / 16;
  dim3 g(So / 4, So / 2, ZT * 2), blk(256, 1, 1);
#define LD(ci, co) if (CI == ci && CO == co) { k_conv_wmma<ci, co, 2, 0, 2, false><<<g, blk, 0, st>>>(in, w, nullptr, 0, 0, mk2, out, os, oo, Sin, Sin, Sin, So, So, So, ZT); return; }
  LD(16, 32) LD(32, 64) LD(64, 128)
#undef LD
}

static void dec_disp(int CI, int CO, const __bf16* in, const unsigned* w,
                     const int* mkF, float* out, int os, int oo, int Sc, hipStream_t st)
{
  int Sf = 2 * Sc;
  int ZT = (Sc + 15) / 16;
  dim3 g(Sc / 4, Sc / 2, ZT * 8 * 2), blk(256, 1, 1);
#define LU(ci, co) if (CI == ci && CO == co) { k_conv_wmma<ci, co, 1, 0, 1, true><<<g, blk, 0, st>>>(in, w, nullptr, 0, 0, mkF, out, os, oo, Sc, Sc, Sc, Sf, Sf, Sf, ZT); return; }
  LU(32, 16) LU(64, 32) LU(128, 64)
#undef LU
}

struct Ctx {
  LvlP* lv;
  float** XIN; float** CC; int** MK;
  float* T1; __bf16* TB;
  unsigned **W1, **W2, **Wd, **Wu, **P1, **P2;
  hipStream_t st;
};

static void bnrelu_h(Ctx& c, const float* x, int xs, int xo, const BNp* bn,
                     const int* mk, int C, int CP, long nv, int relu)
{
  k_bnrelu_bf16<<<nblk(nv * CP), 256, 0, c.st>>>(
      x, xs, xo,
      bn ? bn->g : nullptr, bn ? bn->b : nullptr,
      bn ? bn->mu : nullptr, bn ? bn->v : nullptr,
      mk, c.TB, C, CP, nv, relu);
}

static void run_level(Ctx& c, int l)
{
  int S = 96 >> l, a = 16 << l;
  long nv = 2L * S * S * S;
  const int* mk = c.MK[l];
  bool has = (l < 3);
  float* bo = has ? c.CC[l] : c.XIN[l];
  int bos = has ? 2 * a : a;
  LvlP& L = c.lv[l];

  // entry block: out = x + conv2(bnrelu(conv1(bnrelu(x))*m))*m
  bnrelu_h(c, c.XIN[l], a, 0, &L.blk.bn1, mk, a, a, nv, 1);
  conv3_disp(a, a, c.TB, c.W1[l], nullptr, 0, 0, mk, c.T1, a, 0, S, c.st);
  bnrelu_h(c, c.T1, a, 0, &L.blk.bn2, mk, a, a, nv, 1);
  conv3_disp(a, a, c.TB, c.W2[l], c.XIN[l], a, 0, mk, bo, bos, 0, S, c.st);

  if (has) {
    int b2 = 2 * a, S2 = S / 2;
    long nv2 = 2L * S2 * S2 * S2;
    k_downmask<<<nblk(nv2), 256, 0, c.st>>>(mk, c.MK[l + 1], S2, S2, S2);
    // down
    bnrelu_h(c, c.CC[l], 2 * a, 0, &L.bnd, mk, a, a, nv, 1);
    down_disp(a, b2, c.TB, c.Wd[l], c.MK[l + 1], c.XIN[l + 1], b2, 0, S, c.st);
    // inner
    run_level(c, l + 1);
    // up (deconv) into second half of concat buffer
    bnrelu_h(c, c.XIN[l + 1], b2, 0, &L.bnu, c.MK[l + 1], b2, b2, nv2, 1);
    dec_disp(b2, a, c.TB, c.Wu[l], mk, c.CC[l], 2 * a, a, S2, c.st);
    // post block on concat (2a -> a), identity via NiN
    bnrelu_h(c, c.CC[l], 2 * a, 0, &L.post.bn1, mk, 2 * a, 2 * a, nv, 1);
    conv3_disp(2 * a, a, c.TB, c.P1[l], nullptr, 0, 0, mk, c.T1, a, 0, S, c.st);
    bnrelu_h(c, c.T1, a, 0, &L.post.bn2, mk, a, a, nv, 1);
    k_nin<<<nblk(nv * a), 256, 0, c.st>>>(c.CC[l], 2 * a, 0, L.post.nin,
                                          c.XIN[l], a, 0, 2 * a, a, nv);
    conv3_disp(a, a, c.TB, c.P2[l], c.XIN[l], a, 0, mk, c.XIN[l], a, 0, S, c.st);
  }
}

extern "C" void kernel_launch(void* const* d_in, const int* in_sizes, int n_in,
                              void* d_out, int out_size, void* d_ws, size_t ws_size,
                              hipStream_t stream)
{
  const float* feat = (const float*)d_in[0];
  const int* mask0 = (const int*)d_in[1];

  // JAX pytree order: bn_out(4), lin_b, lin_w, stem, unet{...}
  int cur = 2;
  BNp bn_out;
  bn_out.g = (const float*)d_in[cur++]; bn_out.b = (const float*)d_in[cur++];
  bn_out.mu = (const float*)d_in[cur++]; bn_out.v = (const float*)d_in[cur++];
  const float* lin_b = (const float*)d_in[cur++];
  const float* lin_w = (const float*)d_in[cur++];
  const float* stemw = (const float*)d_in[cur++];
  LvlP lv[4];
  parse_level(0, cur, d_in, lv);

  // ---- workspace bump allocator ----
  size_t off = 0;
  char* base = (char*)d_ws;
  auto alc = [&](size_t bytes) -> void* {
    void* p = base + off;
    off = (off + bytes + 255) & ~(size_t)255;
    return p;
  };

  // ---- repack conv weights to bf16 [tap][ci/2][co][2] ----
  auto repack = [&](const float* w, int ntin, int ntout, int ci, int cipad, int co) -> unsigned* {
    __bf16* o = (__bf16*)alc((size_t)ntout * cipad * co * 2);
    long tot = (long)ntout * cipad * co;
    k_repack<<<nblk(tot), 256, 0, stream>>>(w, o, ntin, ntout, ci, cipad, co);
    return (unsigned*)o;
  };

  unsigned* Wstem = repack(stemw, 27, 28, 3, 16, 16);
  unsigned *W1[4], *W2[4], *Wd[3], *Wu[3], *P1[3], *P2[3];
  for (int l = 0; l < 4; l++) {
    int a = 16 << l;
    W1[l] = repack(lv[l].blk.w1, 27, 28, a, a, a);
    W2[l] = repack(lv[l].blk.w2, 27, 28, a, a, a);
    if (l < 3) {
      int b2 = 2 * a;
      Wd[l] = repack(lv[l].wd, 8, 8, a, a, b2);
      Wu[l] = repack(lv[l].wu, 8, 8, b2, b2, a);
      P1[l] = repack(lv[l].post.w1, 27, 28, b2, b2, a);
      P2[l] = repack(lv[l].post.w2, 27, 28, a, a, a);
    }
  }

  // ---- activations ----
  float* XIN[4]; float* CC[3]; int* MK[4];
  long nv[4]; int S[4];
  for (int l = 0; l < 4; l++) {
    S[l] = 96 >> l;
    nv[l] = 2L * S[l] * S[l] * S[l];
    XIN[l] = (float*)alc((size_t)nv[l] * (16 << l) * 4);
    if (l < 3) CC[l] = (float*)alc((size_t)nv[l] * (32 << l) * 4);
  }
  MK[0] = (int*)mask0;
  for (int l = 1; l < 4; l++) MK[l] = (int*)alc((size_t)nv[l] * 4);
  float* T1 = (float*)alc((size_t)nv[0] * 16 * 4);
  __bf16* TB = (__bf16*)alc((size_t)nv[0] * 32 * 2);

  Ctx c;
  c.lv = lv; c.XIN = XIN; c.CC = CC; c.MK = MK;
  c.T1 = T1; c.TB = TB;
  c.W1 = W1; c.W2 = W2; c.Wd = Wd; c.Wu = Wu; c.P1 = P1; c.P2 = P2;
  c.st = stream;

  // ---- stem: (feat*m, padded to 16ch bf16) -> 3^3 conv -> xin0 ----
  k_bnrelu_bf16<<<nblk(nv[0] * 16), 256, 0, stream>>>(
      feat, 3, 0, nullptr, nullptr, nullptr, nullptr, mask0, TB, 3, 16, nv[0], 0);
  conv3_disp(16, 16, TB, Wstem, nullptr, 0, 0, mask0, XIN[0], 16, 0, 96, stream);

  // ---- UNet ----
  run_level(c, 0);

  // ---- head ----
  k_head<<<nblk(nv[0]), 256, 0, stream>>>(
      XIN[0], bn_out.g, bn_out.b, bn_out.mu, bn_out.v,
      lin_w, lin_b, mask0, (float*)d_out, nv[0]);
}
